// ManualLoRALayer_87170656240352
// MI455X (gfx1250) — matrix operand order repaired and checked by reading, not compiled, not run on hardware
//
#include <hip/hip_runtime.h>
#include <hip/hip_bf16.h>

// ---------------------------------------------------------------------------
// LoRA forward on MI455X (gfx1250):
//   Y = x @ W_base^T + SCALING * (x @ A^T) @ B^T
// Strategy: merge W_eff = W_base + SCALING*B@A (bf16, in workspace), then one
// bf16 WMMA GEMM Y[M,N] = x[M,K] * W_eff[N,K]^T with f32 accumulation.
// M=16384, N=2048, K=2048, R=16.  ~137 GFLOP, ~280 MB HBM -> memory-bound,
// so the inner loop must stay lean: WMMA + b128 moves + packed bf16 converts.
// ---------------------------------------------------------------------------

typedef __attribute__((ext_vector_type(16))) __bf16 v16bf;
typedef __attribute__((ext_vector_type(8)))  __bf16 v8bf;
typedef __attribute__((ext_vector_type(8)))  float  v8f;
typedef __attribute__((ext_vector_type(8)))  float  f32x8;
typedef __attribute__((ext_vector_type(4)))  float  f32x4;
typedef __attribute__((ext_vector_type(4)))  unsigned int u32x4;
typedef __attribute__((ext_vector_type(4)))  int    i32x4;

#define D_IN    2048
#define D_OUT   2048
#define RANK    16
#define SCALING 2.0f   // alpha/rank = 32/16

#define BM   128
#define BN   128
#define BK   32        // one v_wmma_f32_16x16x32_bf16 K-chunk
#define LDK  40        // padded LDS row stride in bf16 elems (80B: conflict-free)
#define TILE_ELEMS (BM * LDK)   // 5120 bf16 per operand tile
#define NK   (D_IN / BK)        // 64 K-steps

// ---- optional gfx1250 async global->LDS path (guarded; safe fallback) -----
#ifndef HAVE_ASYNC
#if defined(__has_builtin)
#if __has_builtin(__builtin_amdgcn_global_load_async_to_lds_b128) && \
    __has_builtin(__builtin_amdgcn_s_wait_asynccnt)
#define HAVE_ASYNC 1
#endif
#endif
#endif
#ifndef HAVE_ASYNC
#define HAVE_ASYNC 0
#endif

// Builtin params are v4i32 pointers (per clang diagnostic). Address-space-
// qualified casts convert implicitly to generic if needed, or match exactly.
typedef __attribute__((address_space(1))) i32x4 gbl_i32x4;
typedef __attribute__((address_space(3))) i32x4 lds_i32x4;

#if HAVE_ASYNC
#define ASYNC_B128(gsrc, ldst)                                   \
  __builtin_amdgcn_global_load_async_to_lds_b128(                \
      (gbl_i32x4*)(gsrc), (lds_i32x4*)(ldst), 0, 0)
#endif

// packed f32x8 -> bf16x8 (backend picks v_cvt_pk_bf16_f32), viewed as 4 dwords
union Pack8 { v8bf bf; u32x4 u; };
__device__ __forceinline__ u32x4 cvt8(f32x4 a, f32x4 b) {
  Pack8 p;
  p.bf = __builtin_convertvector(
      (f32x8){a.x, a.y, a.z, a.w, b.x, b.y, b.z, b.w}, v8bf);
  return p.u;
}

// ---------------------------------------------------------------------------
// Kernel 1: W_eff[o][d] = bf16( W_base[o][d] + SCALING * sum_r B[o][r]*A[r][d] )
// grid (D_IN/256, D_OUT), block 256. A (32K elems) and B stay L2/L0 resident.
// ---------------------------------------------------------------------------
__global__ void lora_merge_w(const float* __restrict__ Wb,
                             const float* __restrict__ A,
                             const float* __restrict__ Bl,
                             __bf16* __restrict__ Weff) {
  const int d = blockIdx.x * blockDim.x + threadIdx.x;
  const int o = blockIdx.y;
  float acc = Wb[(size_t)o * D_IN + d];
#pragma unroll
  for (int r = 0; r < RANK; ++r)
    acc += SCALING * Bl[o * RANK + r] * A[r * D_IN + d];
  Weff[(size_t)o * D_IN + d] = (__bf16)acc;
}

// ---------------------------------------------------------------------------
// Kernel 2: Y = x * W_eff^T.  128x128 block tile, 8 waves (2M x 4N),
// each wave: 4x2 tiles of 16x16 -> 8 WMMA per 32-wide K step.
// Double-buffered LDS, one barrier per K step. x converts f32->bf16 in regs;
// W_eff tile streams via async global->LDS when the toolchain exposes it.
// ---------------------------------------------------------------------------
union FragU { v16bf v; v8bf h[2]; };

__global__ __launch_bounds__(256)
void lora_gemm(const float* __restrict__ X,
               const __bf16* __restrict__ Weff,
               float* __restrict__ Y) {
  __shared__ __bf16 smem[2 * 2 * TILE_ELEMS];   // [buf][A|B][128*40]

  const int tid  = threadIdx.x;
  const int lane = tid & 31;
  const int wave = tid >> 5;
  const int wm   = wave & 1;   // M wave offset: 64*wm
  const int wn   = wave >> 1;  // N wave offset: 32*wn

  const int n0 = blockIdx.x * BN;
  const int m0 = blockIdx.y * BM;

  // per-thread staging coordinates: 512 chunks of 8 elems per 128x32 tile
  const int c0 = tid, c1 = tid + 256;
  const int ar0 = c0 >> 2, ak0 = (c0 & 3) * 8;
  const int ar1 = c1 >> 2, ak1 = (c1 & 3) * 8;

  const float*  xr0 = X    + (size_t)(m0 + ar0) * D_IN + ak0;
  const float*  xr1 = X    + (size_t)(m0 + ar1) * D_IN + ak1;
  const __bf16* wr0 = Weff + (size_t)(n0 + ar0) * D_IN + ak0;
  const __bf16* wr1 = Weff + (size_t)(n0 + ar1) * D_IN + ak1;

  // accumulators: 4 (M) x 2 (N) tiles of 16x16 f32 -> 64 VGPRs
  v8f acc[4][2];
#pragma unroll
  for (int mt = 0; mt < 4; ++mt)
#pragma unroll
    for (int nt = 0; nt < 2; ++nt)
      acc[mt][nt] = (v8f){0.f, 0.f, 0.f, 0.f, 0.f, 0.f, 0.f, 0.f};

  // fragment addressing (ISA 7.12.2 bf16 layouts)
  const int arow = wm * 64 + (lane & 15);      // A: lane -> M row
  const int kg8  = (lane >> 4) * 8;            // A: halves at K = kg8, kg8+16
  const int bcol = wn * 32 + (lane & 15);      // B: lane -> N col
  const int kb   = (lane >> 4) * 8;            // B: halves at K = kb, kb+16

  u32x4 ra0, ra1;            // A-tile staging (converted bf16)
#if !HAVE_ASYNC
  u32x4 rb0, rb1;            // B-tile staging (sync fallback)
#endif

  // ---- prologue: fetch K tile 0 into buffer 0 ----
#if HAVE_ASYNC
  {
    __bf16* Bs = smem + TILE_ELEMS;
    ASYNC_B128(wr0, Bs + ar0 * LDK + ak0);
    ASYNC_B128(wr1, Bs + ar1 * LDK + ak1);
  }
#else
  rb0 = *(const u32x4*)(wr0);
  rb1 = *(const u32x4*)(wr1);
#endif
  ra0 = cvt8(*(const f32x4*)(xr0), *(const f32x4*)(xr0 + 4));
  ra1 = cvt8(*(const f32x4*)(xr1), *(const f32x4*)(xr1 + 4));
  {
    __bf16* As = smem;
    *(u32x4*)(As + ar0 * LDK + ak0) = ra0;
    *(u32x4*)(As + ar1 * LDK + ak1) = ra1;
#if !HAVE_ASYNC
    __bf16* Bs = As + TILE_ELEMS;
    *(u32x4*)(Bs + ar0 * LDK + ak0) = rb0;
    *(u32x4*)(Bs + ar1 * LDK + ak1) = rb1;
#endif
  }
#if HAVE_ASYNC
  __builtin_amdgcn_s_wait_asynccnt(0);
#endif
  __syncthreads();

  int cur = 0;
  for (int kt = 0; kt < NK; ++kt) {
    const bool more = (kt + 1 < NK);
    const int  ko   = (kt + 1) * BK;

    // ---- issue next-tile fetches (overlap with WMMA below) ----
    if (more) {
#if HAVE_ASYNC
      __bf16* Bn = smem + (cur ^ 1) * (2 * TILE_ELEMS) + TILE_ELEMS;
      ASYNC_B128(wr0 + ko, Bn + ar0 * LDK + ak0);
      ASYNC_B128(wr1 + ko, Bn + ar1 * LDK + ak1);
#else
      rb0 = *(const u32x4*)(wr0 + ko);
      rb1 = *(const u32x4*)(wr1 + ko);
#endif
      ra0 = cvt8(*(const f32x4*)(xr0 + ko), *(const f32x4*)(xr0 + ko + 4));
      ra1 = cvt8(*(const f32x4*)(xr1 + ko), *(const f32x4*)(xr1 + ko + 4));
      if (kt + 2 < NK) {                       // gfx1250 global_prefetch_b8
        __builtin_prefetch(xr0 + (kt + 2) * BK, 0, 3);
        __builtin_prefetch(xr1 + (kt + 2) * BK, 0, 3);
      }
    }

    // ---- compute on buffer `cur` ----
    const __bf16* Ac = smem + cur * (2 * TILE_ELEMS);
    const __bf16* Bc = Ac + TILE_ELEMS;

    FragU af[4], bf[2];
#pragma unroll
    for (int mt = 0; mt < 4; ++mt) {
      const __bf16* p = Ac + (arow + mt * 16) * LDK;
      af[mt].h[0] = *(const v8bf*)(p + kg8);        // K = kg8 .. kg8+7
      af[mt].h[1] = *(const v8bf*)(p + kg8 + 16);   // K = kg8+16 .. kg8+23
    }
#pragma unroll
    for (int nt = 0; nt < 2; ++nt) {
      const __bf16* p = Bc + (bcol + nt * 16) * LDK + kb;
      bf[nt].h[0] = *(const v8bf*)(p);              // K = kb .. kb+7
      bf[nt].h[1] = *(const v8bf*)(p + 16);         // K = kb+16 .. kb+23
    }
#pragma unroll
    for (int mt = 0; mt < 4; ++mt)
#pragma unroll
      for (int nt = 0; nt < 2; ++nt)
        acc[mt][nt] = __builtin_amdgcn_wmma_f32_16x16x32_bf16(
            false, af[mt].v, false, bf[nt].v,
            (short)0, acc[mt][nt], false, false);

    // ---- commit next A tile (and B on sync path) into the other buffer ----
    if (more) {
      __bf16* An = smem + (cur ^ 1) * (2 * TILE_ELEMS);
      *(u32x4*)(An + ar0 * LDK + ak0) = ra0;
      *(u32x4*)(An + ar1 * LDK + ak1) = ra1;
#if !HAVE_ASYNC
      __bf16* Bn = An + TILE_ELEMS;
      *(u32x4*)(Bn + ar0 * LDK + ak0) = rb0;
      *(u32x4*)(Bn + ar1 * LDK + ak1) = rb1;
#endif
#if HAVE_ASYNC
      __builtin_amdgcn_s_wait_asynccnt(0);
#endif
    }
    __syncthreads();
    cur ^= 1;
  }

  // ---- epilogue: C/D layout = VGPR j -> M=j (lanes 0-15) / j+8 (16-31),
  //                N = lane%16 ----
  const int col0    = n0 + wn * 32 + (lane & 15);
  const int row_off = (lane >> 4) * 8;
#pragma unroll
  for (int mt = 0; mt < 4; ++mt) {
#pragma unroll
    for (int nt = 0; nt < 2; ++nt) {
      const int col   = col0 + nt * 16;
      const int rbase = m0 + wm * 64 + mt * 16 + row_off;
#pragma unroll
      for (int j = 0; j < 8; ++j)
        Y[(size_t)(rbase + j) * D_OUT + col] = acc[mt][nt][j];
    }
  }
}

// ---------------------------------------------------------------------------
extern "C" void kernel_launch(void* const* d_in, const int* in_sizes, int n_in,
                              void* d_out, int out_size, void* d_ws, size_t ws_size,
                              hipStream_t stream) {
  const float* x  = (const float*)d_in[0];   // [4,4096,2048] f32
  const float* Wb = (const float*)d_in[1];   // [2048,2048]   f32
  const float* A  = (const float*)d_in[2];   // [16,2048]     f32
  const float* Bl = (const float*)d_in[3];   // [2048,16]     f32
  float* Y = (float*)d_out;                  // [4,4096,2048] f32

  __bf16* Weff = (__bf16*)d_ws;              // 2048*2048 bf16 = 8 MiB

  const int M = in_sizes[0] / D_IN;          // 16384

  dim3 mg(D_IN / 256, D_OUT);
  lora_merge_w<<<mg, dim3(256), 0, stream>>>(Wb, A, Bl, Weff);

  dim3 gg(D_OUT / BN, M / BM);               // (16, 128)
  lora_gemm<<<gg, dim3(256), 0, stream>>>(x, Weff, Y);
}
